// DecoderLayer_56908316672219
// MI455X (gfx1250) — matrix operand order repaired and checked by reading, not compiled, not run on hardware
//
#include <hip/hip_runtime.h>
#include <hip/hip_bf16.h>

typedef __bf16 bf16;
typedef __attribute__((ext_vector_type(16))) __bf16 v16bf;
typedef __attribute__((ext_vector_type(8)))  __bf16 bf16x8;
typedef __attribute__((ext_vector_type(8)))  float  v8f;
typedef long long ll;

#define TBM 256
#define TBN 128
#define TBK 32
#define ASTR 40   // TBK + 8 pad: keeps 16B alignment, spreads LDS banks
#define BSTR 40

// ---------------------------------------------------------------------------
// Elementwise / conversion kernels
// ---------------------------------------------------------------------------
__global__ __launch_bounds__(256)
void cvt_f32_bf16_k(const float* __restrict__ s, bf16* __restrict__ d, ll n) {
  ll stride = (ll)gridDim.x * 256;
  for (ll i = (ll)blockIdx.x * 256 + threadIdx.x; i < n; i += stride)
    d[i] = (bf16)s[i];
}

__global__ __launch_bounds__(256)
void concat_cvt_k(const float* __restrict__ member, const float* __restrict__ w,
                  bf16* __restrict__ cat, int ML, int Q, int E, ll n) {
  ll KLE = (ll)(ML + Q) * E;
  ll stride = (ll)gridDim.x * 256;
  for (ll i = (ll)blockIdx.x * 256 + threadIdx.x; i < n; i += stride) {
    ll b = i / KLE; ll rem = i - b * KLE;
    int row = (int)(rem / E); int col = (int)(rem - (ll)row * E);
    float v = (row < ML) ? member[(b * ML + row) * (ll)E + col]
                         : w[(b * Q + (row - ML)) * (ll)E + col];
    cat[i] = (bf16)v;
  }
}

__global__ __launch_bounds__(256)
void qbias_cvt_k(const float* __restrict__ q, const float* __restrict__ rw,
                 const float* __restrict__ rr, bf16* __restrict__ qw,
                 bf16* __restrict__ qr, int HD, ll n) {
  ll stride = (ll)gridDim.x * 256;
  for (ll i = (ll)blockIdx.x * 256 + threadIdx.x; i < n; i += stride) {
    int c = (int)(i % HD);
    float v = q[i];
    qw[i] = (bf16)(v + rw[c]);
    qr[i] = (bf16)(v + rr[c]);
  }
}

// ---------------------------------------------------------------------------
// Tiled bf16 WMMA GEMM:  C[M,N] = A[M,K] * B[K,N] (+epilogue)
// Contract (all launch sites satisfy): M % 256 == 0, K % 32 == 0, N % 16 == 0.
// Block tile 256x128, 8 waves in 4x2 grid, 64x64 per wave = 16 WMMAs/tile.
// Batched over z = b*Hdim + h with independent (b,h) strides per operand.
// Template params: TRANSB (B elem (k,n) at Bm[n*ldb+k]), RELU, OBF (bf16 out).
// Double-buffered LDS pipeline; B tile stored n-major for contiguous frags.
// ---------------------------------------------------------------------------
template <int TRANSB, bool RELU, bool OBF>
__global__ __launch_bounds__(256)
void gemm_bf16_wmma(const bf16* __restrict__ A, ll sAb, ll sAh, int lda,
                    const bf16* __restrict__ Bm, ll sBb, ll sBh, int ldb,
                    void* __restrict__ C, ll sCb, ll sCh, int ldc,
                    int M, int N, int K, int Hdim) {
  __shared__ bf16 As[2][TBM * ASTR];   // 2 x 20 KB
  __shared__ bf16 Bs[2][TBN * BSTR];   // 2 x 10 KB

  const int z = blockIdx.z;
  const int b = z / Hdim, h = z % Hdim;
  A  += b * sAb + h * sAh;
  Bm += b * sBb + h * sBh;
  char* Cp = (char*)C + (b * sCb + h * sCh) * (OBF ? 2 : 4);

  const int tid  = threadIdx.x;
  const int wid  = tid >> 5;
  const int lane = tid & 31;
  const int wm = wid >> 1;            // 4 wave rows (64 rows each)
  const int wn = wid & 1;             // 2 wave cols (64 cols each)
  const int lhalf = lane >> 4;
  const int l15   = lane & 15;

  // --- A staging geometry: one full 32-wide k row per thread ---
  const bf16* aBase = A + (ll)(blockIdx.y * TBM + tid) * lda;

  // --- B staging geometry ---
  const int bnl = tid >> 1;            // TRANSB: local n
  const int bkb = (tid & 1) * 16;      // TRANSB: k sub-offset
  const int gnB = blockIdx.x * TBN + bnl;
  const bool btOk = gnB < N;
  const bf16* btBase = Bm + (ll)gnB * ldb + bkb;

  const int bbr = tid >> 3;            // normal: local k
  const int bbc = (tid & 7) * 16;      // normal: local n base
  const int gnN = blockIdx.x * TBN + bbc;
  const bool bnOk = (gnN + 16) <= N;   // N % 16 == 0 -> all-or-nothing
  const bf16* bnBase = Bm + (ll)bbr * ldb + gnN;

  const int nk = K / TBK;
  bf16x8 ra0, ra1, ra2, ra3, rb0, rb1;
  const bf16x8 zv = {};

  auto gload = [&](int kt) {
    const bf16* ap = aBase + (ll)kt * TBK;
    ra0 = *(const bf16x8*)ap;
    ra1 = *(const bf16x8*)(ap + 8);
    ra2 = *(const bf16x8*)(ap + 16);
    ra3 = *(const bf16x8*)(ap + 24);
    __builtin_prefetch(ap + TBK, 0, 0);                 // global_prefetch_b8
    if constexpr (TRANSB) {
      const bf16* bp = btBase + (ll)kt * TBK;
      rb0 = btOk ? *(const bf16x8*)bp : zv;
      rb1 = btOk ? *(const bf16x8*)(bp + 8) : zv;
      if (btOk) __builtin_prefetch(bp + TBK, 0, 0);
    } else {
      const bf16* bp = bnBase + (ll)kt * TBK * ldb;
      rb0 = bnOk ? *(const bf16x8*)bp : zv;
      rb1 = bnOk ? *(const bf16x8*)(bp + 8) : zv;
      if (bnOk) __builtin_prefetch(bp + (ll)TBK * ldb, 0, 0);
    }
  };

  auto stage = [&](int buf) {
    *(bf16x8*)&As[buf][tid * ASTR]      = ra0;
    *(bf16x8*)&As[buf][tid * ASTR + 8]  = ra1;
    *(bf16x8*)&As[buf][tid * ASTR + 16] = ra2;
    *(bf16x8*)&As[buf][tid * ASTR + 24] = ra3;
    if constexpr (TRANSB) {
      *(bf16x8*)&Bs[buf][bnl * BSTR + bkb]     = rb0;
      *(bf16x8*)&Bs[buf][bnl * BSTR + bkb + 8] = rb1;
    } else {
      #pragma unroll
      for (int e = 0; e < 8; ++e) {
        Bs[buf][(bbc + e) * BSTR + bbr]     = rb0[e];
        Bs[buf][(bbc + 8 + e) * BSTR + bbr] = rb1[e];
      }
    }
  };

  v8f acc[4][4] = {};

  gload(0);
  for (int kt = 0; kt < nk; ++kt) {
    const int buf = kt & 1;
    stage(buf);
    __syncthreads();
    if (kt + 1 < nk) gload(kt + 1);

    // B fragments: k = lhalf*8 .. +7 and 16 + lhalf*8 .. +7 at Bs[n][k]
    v16bf bfrag[4];
    #pragma unroll
    for (int j = 0; j < 4; ++j) {
      const bf16* p = &Bs[buf][(wn * 64 + j * 16 + l15) * BSTR + lhalf * 8];
      bf16x8 lo = *(const bf16x8*)p;
      bf16x8 hi = *(const bf16x8*)(p + 16);
      bfrag[j] = __builtin_shufflevector(lo, hi, 0,1,2,3,4,5,6,7,8,9,10,11,12,13,14,15);
    }
    // A fragments (ISA 7.12.2 16-bit A layout) + 16 WMMAs
    #pragma unroll
    for (int i = 0; i < 4; ++i) {
      const bf16* p = &As[buf][(wm * 64 + i * 16 + l15) * ASTR + lhalf * 8];
      bf16x8 lo = *(const bf16x8*)p;          // k = kb .. kb+7
      bf16x8 hi = *(const bf16x8*)(p + 16);   // k = kb+16 .. kb+23
      v16bf afrag = __builtin_shufflevector(lo, hi, 0,1,2,3,4,5,6,7,8,9,10,11,12,13,14,15);
      #pragma unroll
      for (int j = 0; j < 4; ++j)
        acc[i][j] = __builtin_amdgcn_wmma_f32_16x16x32_bf16(
            false, afrag, false, bfrag[j], (short)0, acc[i][j], false, false);
    }
    __syncthreads();
  }

  // Epilogue: C/D layout -> col = lane&15, row = (lane>>4)*8 + v.
  // M % 256 == 0 by contract -> only the column guard is needed.
  #pragma unroll
  for (int j = 0; j < 4; ++j) {
    const int gn = blockIdx.x * TBN + wn * 64 + j * 16 + l15;
    if (gn >= N) continue;
    #pragma unroll
    for (int i = 0; i < 4; ++i) {
      const ll gm0 = blockIdx.y * TBM + wm * 64 + i * 16 + lhalf * 8;
      #pragma unroll
      for (int v = 0; v < 8; ++v) {
        float val = acc[i][j][v];
        if constexpr (RELU) val = fmaxf(val, 0.0f);
        if constexpr (OBF) ((bf16*)Cp)[(gm0 + v) * ldc + gn] = (bf16)val;
        else               ((float*)Cp)[(gm0 + v) * ldc + gn] = val;
      }
    }
  }
}

// ---------------------------------------------------------------------------
// Softmax with Transformer-XL rel-shift gather + causal mask.
// score(i,j) = (AC[i,j] + BD[i, j + Q-1-i]) * scale  for j <= i+ML, else -inf
// AC row reused as f32 scratch for the exp pass; output bf16 probabilities.
// ---------------------------------------------------------------------------
__global__ __launch_bounds__(256)
void softmax_relshift_k(float* __restrict__ AC, const float* __restrict__ BD,
                        bf16* __restrict__ P, int Q, int KL, int ML, float scale) {
  const int i = blockIdx.x;
  const ll  z = blockIdx.y;
  float* ac = AC + (z * Q + i) * (ll)KL;
  const float* bd = BD + (z * Q + i) * (ll)KL;
  bf16* p = P + (z * Q + i) * (ll)KL;
  const int lim = i + ML;
  const int shift = Q - 1 - i;
  __shared__ float sd[256];
  const int t = threadIdx.x;

  float mx = -3.0e38f;
  for (int j = t; j < KL; j += 256)
    if (j <= lim) mx = fmaxf(mx, (ac[j] + bd[j + shift]) * scale);
  sd[t] = mx; __syncthreads();
  for (int s = 128; s > 0; s >>= 1) { if (t < s) sd[t] = fmaxf(sd[t], sd[t + s]); __syncthreads(); }
  mx = sd[0]; __syncthreads();

  float sum = 0.0f;
  for (int j = t; j < KL; j += 256) {
    float e = 0.0f;
    if (j <= lim) e = __expf((ac[j] + bd[j + shift]) * scale - mx);
    ac[j] = e;
    sum += e;
  }
  sd[t] = sum; __syncthreads();
  for (int s = 128; s > 0; s >>= 1) { if (t < s) sd[t] += sd[t + s]; __syncthreads(); }
  const float inv = 1.0f / sd[0];
  for (int j = t; j < KL; j += 256) p[j] = (bf16)(ac[j] * inv);
}

// ---------------------------------------------------------------------------
// out = LayerNorm(a + resid) * g + beta ; optionally also emit bf16 copy
// ---------------------------------------------------------------------------
__global__ __launch_bounds__(256)
void add_ln_k(const float* __restrict__ a, const float* __restrict__ resid,
              const float* __restrict__ g, const float* __restrict__ bta,
              float* __restrict__ outf, bf16* __restrict__ outb, int E, float eps) {
  const ll row = blockIdx.x;
  const float* ar = a + row * E;
  const float* rr = resid + row * E;
  __shared__ float sd[256];
  const int t = threadIdx.x;

  float ls = 0.0f;
  for (int c = t; c < E; c += 256) ls += ar[c] + rr[c];
  sd[t] = ls; __syncthreads();
  for (int s = 128; s > 0; s >>= 1) { if (t < s) sd[t] += sd[t + s]; __syncthreads(); }
  const float mu = sd[0] / (float)E; __syncthreads();

  float lv = 0.0f;
  for (int c = t; c < E; c += 256) { float d = ar[c] + rr[c] - mu; lv += d * d; }
  sd[t] = lv; __syncthreads();
  for (int s = 128; s > 0; s >>= 1) { if (t < s) sd[t] += sd[t + s]; __syncthreads(); }
  const float rstd = rsqrtf(sd[0] / (float)E + eps);

  for (int c = t; c < E; c += 256) {
    float v = (ar[c] + rr[c] - mu) * rstd * g[c] + bta[c];
    outf[row * E + c] = v;
    if (outb) outb[row * E + c] = (bf16)v;
  }
}

// ---------------------------------------------------------------------------
// Host launcher
// ---------------------------------------------------------------------------
static inline size_t alignUp(size_t x, size_t a) { return (x + a - 1) & ~(a - 1); }

template <int TRANSB, bool RELU, bool OBF>
static void launch_gemm(const bf16* A, ll sAb, ll sAh, int lda,
                        const bf16* Bm, ll sBb, ll sBh, int ldb,
                        void* C, ll sCb, ll sCh, int ldc,
                        int M, int N, int K, int Hdim, int nz, hipStream_t stream) {
  dim3 grid((N + TBN - 1) / TBN, (M + TBM - 1) / TBM, nz);
  gemm_bf16_wmma<TRANSB, RELU, OBF><<<grid, dim3(256), 0, stream>>>(
      A, sAb, sAh, lda, Bm, sBb, sBh, ldb, C, sCb, sCh, ldc, M, N, K, Hdim);
}

extern "C" void kernel_launch(void* const* d_in, const int* in_sizes, int n_in,
                              void* d_out, int out_size, void* d_ws, size_t ws_size,
                              hipStream_t stream) {
  const int Bb = 2, Q = 1024, ML = 1024, E = 1024, H = 16, DH = 64;
  const int KL = Q + ML, HD = H * DH, FF = 4 * E;
  const float scale = 1.0f / 32.0f;   // 1/sqrt(E)
  const float eps = 1e-3f;

  const float* w      = (const float*)d_in[0];
  const float* r      = (const float*)d_in[1];
  const float* member = (const float*)d_in[2];
  const float* Wq  = (const float*)d_in[4];
  const float* Wk  = (const float*)d_in[5];
  const float* Wv  = (const float*)d_in[6];
  const float* Wr  = (const float*)d_in[7];
  const float* Wo  = (const float*)d_in[8];
  const float* rwb = (const float*)d_in[9];
  const float* rrb = (const float*)d_in[10];
  const float* g1  = (const float*)d_in[11];
  const float* b1  = (const float*)d_in[12];
  const float* W1  = (const float*)d_in[13];
  const float* W2  = (const float*)d_in[14];
  const float* g2  = (const float*)d_in[15];
  const float* b2  = (const float*)d_in[16];
  float* out = (float*)d_out;

  char* wsp = (char*)d_ws;
  size_t off = 0;
  auto alloc = [&](size_t bytes) -> void* {
    void* p = wsp + off; off = alignUp(off + bytes, 256); return p;
  };

  bf16* cat_bf = (bf16*)alloc((size_t)Bb * KL * E * 2);
  bf16* r_bf   = (bf16*)alloc((size_t)KL * E * 2);
  bf16* wq_bf  = (bf16*)alloc((size_t)E * HD * 2);
  bf16* wk_bf  = (bf16*)alloc((size_t)E * HD * 2);
  bf16* wv_bf  = (bf16*)alloc((size_t)E * HD * 2);
  bf16* wr_bf  = (bf16*)alloc((size_t)E * HD * 2);
  bf16* wo_bf  = (bf16*)alloc((size_t)HD * E * 2);
  bf16* w1_bf  = (bf16*)alloc((size_t)E * FF * 2);
  bf16* w2_bf  = (bf16*)alloc((size_t)FF * E * 2);
  float* q_f   = (float*)alloc((size_t)Bb * Q * HD * 4);
  bf16* qw_bf  = (bf16*)alloc((size_t)Bb * Q * HD * 2);
  bf16* qr_bf  = (bf16*)alloc((size_t)Bb * Q * HD * 2);
  bf16* k_bf   = (bf16*)alloc((size_t)Bb * KL * HD * 2);
  bf16* v_bf   = (bf16*)alloc((size_t)Bb * KL * HD * 2);
  bf16* rp_bf  = (bf16*)alloc((size_t)KL * HD * 2);
  float* AC    = (float*)alloc((size_t)Bb * H * Q * KL * 4);
  float* BD    = (float*)alloc((size_t)Bb * H * Q * KL * 4);
  bf16* P_bf   = (bf16*)alloc((size_t)Bb * H * Q * KL * 2);
  bf16* o_bf   = (bf16*)alloc((size_t)Bb * Q * HD * 2);
  float* at_f  = (float*)alloc((size_t)Bb * Q * E * 4);
  float* x_f   = (float*)alloc((size_t)Bb * Q * E * 4);
  bf16* x_bf   = (bf16*)alloc((size_t)Bb * Q * E * 2);
  bf16* h_bf   = (bf16*)alloc((size_t)Bb * Q * FF * 2);
  float* y_f   = (float*)alloc((size_t)Bb * Q * E * 4);
  (void)ws_size; (void)in_sizes; (void)n_in; (void)out_size;

  auto cvt = [&](const float* s, bf16* d, ll n) {
    cvt_f32_bf16_k<<<dim3((unsigned)((n + 255) / 256)), dim3(256), 0, stream>>>(s, d, n);
  };
  cvt(Wq, wq_bf, (ll)E * HD);  cvt(Wk, wk_bf, (ll)E * HD);
  cvt(Wv, wv_bf, (ll)E * HD);  cvt(Wr, wr_bf, (ll)E * HD);
  cvt(Wo, wo_bf, (ll)HD * E);  cvt(W1, w1_bf, (ll)E * FF);
  cvt(W2, w2_bf, (ll)FF * E);  cvt(r,  r_bf,  (ll)KL * E);

  {
    ll n = (ll)Bb * KL * E;
    concat_cvt_k<<<dim3((unsigned)((n + 255) / 256)), dim3(256), 0, stream>>>(
        member, w, cat_bf, ML, Q, E, n);
  }

  // q = w @ Wq  (rows MLEN.. of cat), f32 out
  launch_gemm<0, false, false>(cat_bf + (ll)ML * E, (ll)KL * E, 0, E, wq_bf, 0, 0, HD,
                               q_f, (ll)Q * HD, 0, HD, Q, HD, E, 1, Bb, stream);
  // k, v = cat @ Wk/Wv (bf16 out)
  launch_gemm<0, false, true>(cat_bf, (ll)KL * E, 0, E, wk_bf, 0, 0, HD,
                              k_bf, (ll)KL * HD, 0, HD, KL, HD, E, 1, Bb, stream);
  launch_gemm<0, false, true>(cat_bf, (ll)KL * E, 0, E, wv_bf, 0, 0, HD,
                              v_bf, (ll)KL * HD, 0, HD, KL, HD, E, 1, Bb, stream);
  // rp = r @ Wr (bf16 out)
  launch_gemm<0, false, true>(r_bf, 0, 0, E, wr_bf, 0, 0, HD,
                              rp_bf, 0, 0, HD, KL, HD, E, 1, 1, stream);

  {
    ll n = (ll)Bb * Q * HD;
    qbias_cvt_k<<<dim3((unsigned)((n + 255) / 256)), dim3(256), 0, stream>>>(
        q_f, rwb, rrb, qw_bf, qr_bf, HD, n);
  }

  // AC[b,h] = qw @ K^T   [Q x KL], reduction dim = 64
  launch_gemm<1, false, false>(qw_bf, (ll)Q * HD, DH, HD,
                               k_bf, (ll)KL * HD, DH, HD,
                               AC, (ll)H * Q * KL, (ll)Q * KL, KL,
                               Q, KL, DH, H, Bb * H, stream);
  // BDraw[b,h] = qr @ R^T  (R shared across batch)
  launch_gemm<1, false, false>(qr_bf, (ll)Q * HD, DH, HD,
                               rp_bf, 0, DH, HD,
                               BD, (ll)H * Q * KL, (ll)Q * KL, KL,
                               Q, KL, DH, H, Bb * H, stream);

  softmax_relshift_k<<<dim3(Q, Bb * H), dim3(256), 0, stream>>>(
      AC, BD, P_bf, Q, KL, ML, scale);

  // o[b,:,h*64:(h+1)*64] = P[b,h] @ V[b,h]   (bf16 out, strided epilogue)
  launch_gemm<0, false, true>(P_bf, (ll)H * Q * KL, (ll)Q * KL, KL,
                              v_bf, (ll)KL * HD, DH, HD,
                              o_bf, (ll)Q * HD, DH, HD,
                              Q, DH, KL, H, Bb * H, stream);

  // attn_out = o @ Wo  (f32)
  launch_gemm<0, false, false>(o_bf, 0, 0, HD, wo_bf, 0, 0, E,
                               at_f, 0, 0, E, Bb * Q, E, HD, 1, 1, stream);

  // x = LN(w + attn_out)
  add_ln_k<<<dim3(Bb * Q), dim3(256), 0, stream>>>(at_f, w, g1, b1, x_f, x_bf, E, eps);

  // h = relu(x @ W1) (bf16 out), y = h @ W2 (f32)
  launch_gemm<0, true, true>(x_bf, 0, 0, E, w1_bf, 0, 0, FF,
                             h_bf, 0, 0, FF, Bb * Q, FF, E, 1, 1, stream);
  launch_gemm<0, false, false>(h_bf, 0, 0, FF, w2_bf, 0, 0, E,
                               y_f, 0, 0, E, Bb * Q, E, FF, 1, 1, stream);

  // out = LN(y + x)
  add_ln_k<<<dim3(Bb * Q), dim3(256), 0, stream>>>(y_f, x_f, g2, b2, out, (bf16*)nullptr, E, eps);
}
